// AttentionSeqModel_11235634446962
// MI455X (gfx1250) — matrix operand order repaired and checked by reading, not compiled, not run on hardware
//
#include <hip/hip_runtime.h>
#include <hip/hip_bf16.h>

// Sizes from the reference
static constexpr int B = 1024, L = 256, D_OBS = 256, H = 128, A = 2;
static constexpr int NSWAP = 6;
static constexpr int G3 = 3 * H; // 384

typedef __attribute__((ext_vector_type(16))) __bf16 v16bf;
typedef __attribute__((ext_vector_type(8)))  float  v8f;

// CDNA5 async global->LDS copies (ASYNCcnt-tracked), guarded for portability.
#if defined(__has_builtin)
#  if __has_builtin(__builtin_amdgcn_global_load_async_to_lds_b128)
#    define USE_ASYNC_LDS 1
#  endif
#  if __has_builtin(__builtin_amdgcn_s_wait_asynccnt)
#    define HAVE_WAIT_ASYNC_BUILTIN 1
#  endif
#endif

#if defined(USE_ASYNC_LDS)
// Builtin signature (from hipcc diagnostic): param0 is a NON-const pointer to
// int __attribute__((vector_size(16))) in the global (AS1) address space.
typedef int v4i_ __attribute__((vector_size(16)));
typedef __attribute__((address_space(1))) v4i_* gv4i_ptr;
typedef __attribute__((address_space(3))) v4i_* lv4i_ptr;
#endif

__device__ __forceinline__ void wait_asynccnt0() {
#if defined(USE_ASYNC_LDS)
#  if defined(HAVE_WAIT_ASYNC_BUILTIN)
  __builtin_amdgcn_s_wait_asynccnt(0);
#  else
  asm volatile("s_wait_asynccnt 0x0" ::: "memory");
#  endif
#endif
}

// ---------------------------------------------------------------------------
// Helpers
// ---------------------------------------------------------------------------
__device__ __forceinline__ v16bf cvt16_bf16(float4 f0, float4 f1, float4 f2, float4 f3) {
  v16bf t;
  t[0]  = (__bf16)f0.x; t[1]  = (__bf16)f0.y; t[2]  = (__bf16)f0.z; t[3]  = (__bf16)f0.w;
  t[4]  = (__bf16)f1.x; t[5]  = (__bf16)f1.y; t[6]  = (__bf16)f1.z; t[7]  = (__bf16)f1.w;
  t[8]  = (__bf16)f2.x; t[9]  = (__bf16)f2.y; t[10] = (__bf16)f2.z; t[11] = (__bf16)f2.w;
  t[12] = (__bf16)f3.x; t[13] = (__bf16)f3.y; t[14] = (__bf16)f3.z; t[15] = (__bf16)f3.w;
  return t;
}

// A 16x32 fragment: lane m=lane&15; lanes<16 -> K 0..15,
// lanes>=16 -> K 16..31. Row-major LDS tile, two contiguous 16B reads.
__device__ __forceinline__ v16bf load_A_frag(const __bf16* Abuf, int lane) {
  int m  = lane & 15;
  int kb = (lane & 16) ? 16 : 0;
  const __bf16* row = Abuf + m * 32 + kb;
  union { uint4 u[2]; v16bf v; } t;
  t.u[0] = *(const uint4*)(row);
  t.u[1] = *(const uint4*)(row + 8);
  return t.v;
}

// Epilogue modes
#define EPI_F32       0
#define EPI_RELU_F32  1
#define EPI_BF16      2
#define EPI_TANH_BF16 3

// ---------------------------------------------------------------------------
// Tiled WMMA GEMM body: C[M,N] = act(A[M,K] @ W[N,K]^T + bias)
// 128x128 tile, 256 threads (8 waves, 4Mx2N), wave sub-tile 32x64 = 2x4 WMMAs.
// Double-buffered LDS; B staged in WMMA-fragment order so a B fragment is one
// contiguous 32B LDS read. bf16-A staging uses GLOBAL_LOAD_ASYNC_TO_LDS_B128
// when available. AROW_MODE==1 applies the buggy obs row-swap remap.
// ---------------------------------------------------------------------------
template<bool A_BF16, int AROW_MODE, int EPI>
__device__ __forceinline__
void gemm_body(const void* __restrict__ Ap, long lda,
               const float* __restrict__ W, const float* __restrict__ bias,
               float* __restrict__ Cf, __bf16* __restrict__ Cbf, int ldc,
               int M, int N, int K, int ntn, int blockId) {
  __shared__ alignas(32) __bf16 As[2][128 * 32];
  __shared__ alignas(32) __bf16 Bs[2][128 * 32]; // frag order: [(nsub*32+lane)*16+e]

  const int mt = blockId / ntn, nt = blockId % ntn;
  const int m0 = mt * 128, n0 = nt * 128;
  const int tid = threadIdx.x;
  const int wave = tid >> 5, lane = tid & 31;
  const int wm = (wave & 3) * 32;   // wave M offset within tile
  const int wn = (wave >> 2) * 64;  // wave N offset within tile

  // Staging assignment: thread -> (row sm, k-half sk)
  const int sm = tid >> 1;          // 0..127 (A row / W row)
  const int sk = (tid & 1) * 16;    // 0 or 16

  long amg = m0 + sm;               // A source row (with optional swap remap)
  if (AROW_MODE == 1) {
    long bb = amg >> 8;
    int  l  = (int)(amg & 255);
    l = (l < NSWAP) ? (L - 1 - l) : l;
    amg = (bb << 8) | l;
  }
  const char*  Abase = (const char*)Ap;
  const float* Wrow  = W + (long)(n0 + sm) * K;
  // B fragment-order base: laneIdx = sk + (n&15), nsub = n>>4
  const int bBase = (((sm >> 4) * 32) + sk + (sm & 15)) * 16;
  const int aBase = sm * 32 + sk;

  auto stageA = [&](int k0, int buf) {
    if (A_BF16) {
      const char* gsrc = Abase + (amg * lda + k0 + sk) * 2;
#if defined(USE_ASYNC_LDS)
      char* g = const_cast<char*>(gsrc);
      __builtin_amdgcn_global_load_async_to_lds_b128(
          (gv4i_ptr)g, (lv4i_ptr)&As[buf][aBase], 0, 0);
      __builtin_amdgcn_global_load_async_to_lds_b128(
          (gv4i_ptr)g, (lv4i_ptr)&As[buf][aBase], 16, 0);
#else
      const uint4* src = (const uint4*)gsrc;
      uint4 u0 = src[0], u1 = src[1];
      *(uint4*)&As[buf][aBase]     = u0;
      *(uint4*)&As[buf][aBase + 8] = u1;
#endif
    } else {
      const float4* src = (const float4*)(Abase + (amg * lda + k0 + sk) * 4);
      float4 f0 = src[0], f1 = src[1], f2 = src[2], f3 = src[3];
      *(v16bf*)&As[buf][aBase] = cvt16_bf16(f0, f1, f2, f3);
    }
  };
  auto stageB = [&](int k0, int buf) {
    const float4* src = (const float4*)(Wrow + k0 + sk);
    float4 f0 = src[0], f1 = src[1], f2 = src[2], f3 = src[3];
    *(v16bf*)&Bs[buf][bBase] = cvt16_bf16(f0, f1, f2, f3);
  };

  v8f acc[2][4];
#pragma unroll
  for (int i = 0; i < 2; ++i)
#pragma unroll
    for (int j = 0; j < 4; ++j)
#pragma unroll
      for (int v = 0; v < 8; ++v) acc[i][j][v] = 0.0f;

  const int nk = K >> 5;
  stageA(0, 0);
  stageB(0, 0);

  for (int kt = 0; kt < nk; ++kt) {
    if (A_BF16) wait_asynccnt0();  // async LDS writes visible before barrier
    __syncthreads();
    const int cur = kt & 1;
    if (kt + 1 < nk) {
      stageA((kt + 1) * 32, cur ^ 1);
      stageB((kt + 1) * 32, cur ^ 1);
      if (kt + 2 < nk) {  // distant prefetch into L2 (global_prefetch_b8)
        __builtin_prefetch(Wrow + (kt + 2) * 32 + sk, 0, 1);
      }
    }
    v16bf af[2], bv[4];
#pragma unroll
    for (int i = 0; i < 2; ++i) af[i] = load_A_frag(&As[cur][(wm + i * 16) * 32], lane);
#pragma unroll
    for (int j = 0; j < 4; ++j)
      bv[j] = *(const v16bf*)&Bs[cur][(((wn >> 4) + j) * 32 + lane) * 16];
#pragma unroll
    for (int i = 0; i < 2; ++i)
#pragma unroll
      for (int j = 0; j < 4; ++j)
        acc[i][j] = __builtin_amdgcn_wmma_f32_16x16x32_bf16(
            false, af[i], false, bv[j], (short)0, acc[i][j], false, false);
  }

  // Epilogue: C layout lanes 0-15 N=lane,M=v ; lanes 16-31 M=8+v
  const int cn = lane & 15;
  const int mb = (lane & 16) ? 8 : 0;
#pragma unroll
  for (int i = 0; i < 2; ++i)
#pragma unroll
    for (int j = 0; j < 4; ++j)
#pragma unroll
      for (int v = 0; v < 8; ++v) {
        int m = m0 + wm + i * 16 + mb + v;
        int n = n0 + wn + j * 16 + cn;
        if (m < M && n < N) {
          float val = acc[i][j][v] + bias[n];
          if (EPI == EPI_F32)           Cf[(long)m * ldc + n] = val;
          else if (EPI == EPI_RELU_F32) Cf[(long)m * ldc + n] = fmaxf(val, 0.0f);
          else if (EPI == EPI_BF16)     Cbf[(long)m * ldc + n] = (__bf16)val;
          else                          Cbf[(long)m * ldc + n] = (__bf16)tanhf(val);
        }
      }
}

template<bool A_BF16, int AROW_MODE, int EPI>
__global__ __launch_bounds__(256)
void gemm_wmma_kernel(const void* __restrict__ Ap, long lda,
                      const float* __restrict__ W, const float* __restrict__ bias,
                      float* __restrict__ Cf, __bf16* __restrict__ Cbf, int ldc,
                      int M, int N, int K, int ntn) {
  gemm_body<A_BF16, AROW_MODE, EPI>(Ap, lda, W, bias, Cf, Cbf, ldc, M, N, K, ntn, blockIdx.x);
}

// One launch computing both GRU pre-activations: gi = A0@W0^T+b0, gh = A1@W1^T+b1
// grid (24, 2); blockIdx.y selects the operand set (block-uniform branch).
__global__ __launch_bounds__(256)
void gru_dual_gemm_kernel(const __bf16* __restrict__ A0, long lda0,
                          const float* __restrict__ W0, const float* __restrict__ b0,
                          const __bf16* __restrict__ A1, long lda1,
                          const float* __restrict__ W1, const float* __restrict__ b1,
                          float* __restrict__ gi, float* __restrict__ gh) {
  if (blockIdx.y == 0)
    gemm_body<true, 0, EPI_F32>(A0, lda0, W0, b0, gi, nullptr, G3, B, G3, H, 3, blockIdx.x);
  else
    gemm_body<true, 0, EPI_F32>(A1, lda1, W1, b1, gh, nullptr, G3, B, G3, H, 3, blockIdx.x);
}

// ---------------------------------------------------------------------------
// Elementwise / small kernels
// ---------------------------------------------------------------------------
__device__ __forceinline__ float sigmoidf_(float x) { return 1.0f / (1.0f + expf(-x)); }

__global__ void init_h_kernel(float* h_f32, __bf16* h_bf) {
  int idx = blockIdx.x * blockDim.x + threadIdx.x;
  if (idx < B * H) { h_f32[idx] = 0.0f; h_bf[idx] = (__bf16)0.0f; }
}

// GRU gates (torch order r,z,n): h2 = (1-z)*n + z*h ; optionally records enc_out[:,t,:]
__global__ void gru_gates_kernel(const float* __restrict__ gi, const float* __restrict__ gh,
                                 float* __restrict__ h_f32, __bf16* __restrict__ h_bf,
                                 __bf16* __restrict__ enc_out, int t) {
  int idx = blockIdx.x * blockDim.x + threadIdx.x;
  if (idx >= B * H) return;
  int b = idx >> 7, j = idx & 127;
  const float* gib = gi + (long)b * G3;
  const float* ghb = gh + (long)b * G3;
  float r = sigmoidf_(gib[j] + ghb[j]);
  float z = sigmoidf_(gib[j + H] + ghb[j + H]);
  float n = tanhf(gib[j + 2 * H] + r * ghb[j + 2 * H]);
  float h = h_f32[idx];
  float h2 = (1.0f - z) * n + z * h;
  h_f32[idx] = h2;
  __bf16 hb = (__bf16)h2;
  h_bf[idx] = hb;
  if (enc_out) enc_out[((long)b * L + t) * H + j] = hb;
}

// Decoder pre-step: e = y_prev @ dec_emb_W^T + b (K=2), build Acat1=[e,h], Acat2[:H]=e
__global__ void dec_pre_kernel(const float* __restrict__ ybuf,
                               const float* __restrict__ demb_W, const float* __restrict__ demb_b,
                               const __bf16* __restrict__ h_bf,
                               __bf16* __restrict__ Acat1, __bf16* __restrict__ Acat2, int t) {
  int idx = blockIdx.x * blockDim.x + threadIdx.x;
  if (idx >= B * H) return;
  int b = idx >> 7, j = idx & 127;
  float y0 = 0.0f, y1 = 0.0f;
  if (t > 0) { y0 = ybuf[b * 2]; y1 = ybuf[b * 2 + 1]; }
  float e = demb_W[j * 2 + 0] * y0 + demb_W[j * 2 + 1] * y1 + demb_b[j];
  __bf16 eb = (__bf16)e;
  Acat1[(long)b * 256 + j] = eb;
  Acat1[(long)b * 256 + H + j] = h_bf[idx];
  Acat2[(long)b * 256 + j] = eb;
}

// Fused: softmax(logits[b,:]) * attn_mask[t,:], then
// applied[b,j] = sum_l w[l] * enc_bf[b,l,j]  -> Acat2[b, H+j] (bf16).
// 256 threads/block: softmax over L=256, then 64 column-pairs x 4 L-quarters.
// enc rows read as packed uint (2 bf16), bf16->f32 via bit shift.
__global__ __launch_bounds__(256)
void attn_apply_kernel(const float* __restrict__ logits,
                       const float* __restrict__ attn_mask,
                       const __bf16* __restrict__ enc_bf,
                       __bf16* __restrict__ Acat2, int t) {
  __shared__ float sdata[256];
  __shared__ float ws[256];
  __shared__ float part0[4][64];
  __shared__ float part1[4][64];
  const int b = blockIdx.x, tid = threadIdx.x;

  float x = logits[(long)b * L + tid];
  sdata[tid] = x;
  __syncthreads();
  for (int s = 128; s > 0; s >>= 1) {
    if (tid < s) sdata[tid] = fmaxf(sdata[tid], sdata[tid + s]);
    __syncthreads();
  }
  float mx = sdata[0];
  __syncthreads();
  float ex = expf(x - mx);
  sdata[tid] = ex;
  __syncthreads();
  for (int s = 128; s > 0; s >>= 1) {
    if (tid < s) sdata[tid] += sdata[tid + s];
    __syncthreads();
  }
  float sum = sdata[0];
  ws[tid] = (ex / sum) * attn_mask[(long)t * L + tid];
  __syncthreads();

  const int j2 = tid & 63;  // column pair: columns 2*j2, 2*j2+1
  const int lq = tid >> 6;  // L quarter
  const unsigned int* Eu = (const unsigned int*)(enc_bf + (long)b * L * H);
  float a0 = 0.0f, a1 = 0.0f;
  const int l0 = lq * 64;
#pragma unroll 8
  for (int l = l0; l < l0 + 64; ++l) {
    unsigned int u = Eu[l * (H / 2) + j2];
    float f0 = __uint_as_float(u << 16);
    float f1 = __uint_as_float(u & 0xffff0000u);
    float wl = ws[l];
    a0 += wl * f0;
    a1 += wl * f1;
  }
  part0[lq][j2] = a0;
  part1[lq][j2] = a1;
  __syncthreads();
  if (tid < 64) {
    float s0 = part0[0][tid] + part0[1][tid] + part0[2][tid] + part0[3][tid];
    float s1 = part1[0][tid] + part1[1][tid] + part1[2][tid] + part1[3][tid];
    Acat2[(long)b * 256 + H + 2 * tid]     = (__bf16)s0;
    Acat2[(long)b * 256 + H + 2 * tid + 1] = (__bf16)s1;
  }
}

// y = tanh(h2 @ out_W^T + out_b), feedback + final-output stores with reverse swap
__global__ void dec_out_kernel(const float* __restrict__ h_f32,
                               const float* __restrict__ out_W, const float* __restrict__ out_b,
                               float* __restrict__ ybuf, float* __restrict__ outp, int t) {
  int wave = threadIdx.x >> 5, lane = threadIdx.x & 31;
  int b = blockIdx.x * 8 + wave;
  const float* hr = h_f32 + (long)b * H;
  float a0 = 0.0f, a1 = 0.0f;
  for (int k = lane; k < H; k += 32) {
    float hv = hr[k];
    a0 += hv * out_W[k];
    a1 += hv * out_W[H + k];
  }
  for (int off = 16; off > 0; off >>= 1) {
    a0 += __shfl_down(a0, off, 32);
    a1 += __shfl_down(a1, off, 32);
  }
  if (lane == 0) {
    float y0 = tanhf(a0 + out_b[0]);
    float y1 = tanhf(a1 + out_b[1]);
    ybuf[b * 2] = y0; ybuf[b * 2 + 1] = y1;
    if (t >= NSWAP) {
      outp[(long)b * (L * A) + t * 2]     = y0;
      outp[(long)b * (L * A) + t * 2 + 1] = y1;
    }
    if (t >= L - NSWAP) {
      int r = L - 1 - t;
      outp[(long)b * (L * A) + r * 2]     = y0;
      outp[(long)b * (L * A) + r * 2 + 1] = y1;
    }
  }
}

// value[b] = v2[b,:] . v3_W + v3_b
__global__ void value_out_kernel(const float* __restrict__ v2buf,
                                 const float* __restrict__ v3_W, const float* __restrict__ v3_b,
                                 float* __restrict__ valp) {
  int wave = threadIdx.x >> 5, lane = threadIdx.x & 31;
  int b = blockIdx.x * 8 + wave;
  const float* r = v2buf + (long)b * 128;
  float a = 0.0f;
  for (int k = lane; k < 128; k += 32) a += r[k] * v3_W[k];
  for (int off = 16; off > 0; off >>= 1) a += __shfl_down(a, off, 32);
  if (lane == 0) valp[b] = a + v3_b[0];
}

// ---------------------------------------------------------------------------
extern "C" void kernel_launch(void* const* d_in, const int* in_sizes, int n_in,
                              void* d_out, int out_size, void* d_ws, size_t ws_size,
                              hipStream_t stream) {
  const float* obs       = (const float*)d_in[0];
  const float* attn_mask = (const float*)d_in[1];
  const float* enc_emb_W = (const float*)d_in[2];
  const float* enc_emb_b = (const float*)d_in[3];
  const float* enc_Wih   = (const float*)d_in[4];
  const float* enc_Whh   = (const float*)d_in[5];
  const float* enc_bih   = (const float*)d_in[6];
  const float* enc_bhh   = (const float*)d_in[7];
  const float* dec_emb_W = (const float*)d_in[8];
  const float* dec_emb_b = (const float*)d_in[9];
  const float* attn_W    = (const float*)d_in[10];
  const float* attn_b    = (const float*)d_in[11];
  const float* comb_W    = (const float*)d_in[12];
  const float* comb_b    = (const float*)d_in[13];
  const float* dec_Wih   = (const float*)d_in[14];
  const float* dec_Whh   = (const float*)d_in[15];
  const float* dec_bih   = (const float*)d_in[16];
  const float* dec_bhh   = (const float*)d_in[17];
  const float* out_W     = (const float*)d_in[18];
  const float* out_b     = (const float*)d_in[19];
  const float* v1_W      = (const float*)d_in[20];
  const float* v1_b      = (const float*)d_in[21];
  const float* v2_W      = (const float*)d_in[22];
  const float* v2_b      = (const float*)d_in[23];
  const float* v3_W      = (const float*)d_in[24];
  const float* v3_b      = (const float*)d_in[25];
  (void)in_sizes; (void)n_in; (void)out_size; (void)ws_size;

  char* ws = (char*)d_ws;
  auto alloc = [&](size_t bytes) -> char* {
    char* p = ws;
    ws += (bytes + 255) & ~(size_t)255;
    return p;
  };
  __bf16* emb_bf = (__bf16*)alloc((size_t)B * L * H * 2); // encoder embeddings (bf16)
  __bf16* enc_bf = (__bf16*)alloc((size_t)B * L * H * 2); // encoder outputs (bf16)
  float*  h_f32  = (float*)alloc((size_t)B * H * 4);
  __bf16* h_bf   = (__bf16*)alloc((size_t)B * H * 2);
  float*  gi     = (float*)alloc((size_t)B * G3 * 4);
  float*  gh     = (float*)alloc((size_t)B * G3 * 4);
  float*  v1buf  = (float*)alloc((size_t)B * 128 * 4);
  float*  v2buf  = (float*)alloc((size_t)B * 128 * 4);
  __bf16* Acat1  = (__bf16*)alloc((size_t)B * 256 * 2);  // [e, h]
  __bf16* Acat2  = (__bf16*)alloc((size_t)B * 256 * 2);  // [e, applied]
  float*  logits = (float*)alloc((size_t)B * L * 4);
  __bf16* g_bf   = (__bf16*)alloc((size_t)B * H * 2);
  float*  ybuf   = (float*)alloc((size_t)B * A * 4);

  float* outp = (float*)d_out;             // [B, L*A]
  float* valp = outp + (size_t)B * L * A;  // [B, 1]

  // ---- value branch (un-swapped obs): two relu GEMMs + dot ----
  gemm_wmma_kernel<false, 0, EPI_RELU_F32><<<8, 256, 0, stream>>>(
      obs, (long)L * D_OBS, v1_W, v1_b, v1buf, nullptr, 128, B, 128, L * D_OBS, 1);
  gemm_wmma_kernel<false, 0, EPI_RELU_F32><<<8, 256, 0, stream>>>(
      v1buf, 128, v2_W, v2_b, v2buf, nullptr, 128, B, 128, 128, 1);
  value_out_kernel<<<B / 8, 256, 0, stream>>>(v2buf, v3_W, v3_b, valp);

  // ---- encoder embedding with buggy row-swap remap (A-row remap) ----
  gemm_wmma_kernel<false, 1, EPI_BF16><<<(B * L) / 128, 256, 0, stream>>>(
      obs, (long)D_OBS, enc_emb_W, enc_emb_b, nullptr, emb_bf, H, B * L, H, D_OBS, 1);

  init_h_kernel<<<(B * H) / 256, 256, 0, stream>>>(h_f32, h_bf);

  // ---- encoder GRU scan ----
  for (int t = 0; t < L; ++t) {
    gru_dual_gemm_kernel<<<dim3(24, 2), 256, 0, stream>>>(
        emb_bf + (size_t)t * H, (long)L * H, enc_Wih, enc_bih,
        h_bf, (long)H, enc_Whh, enc_bhh, gi, gh);
    gru_gates_kernel<<<(B * H) / 256, 256, 0, stream>>>(gi, gh, h_f32, h_bf, enc_bf, t);
  }

  // ---- decoder scan (h starts at final encoder h; y_prev starts at 0) ----
  for (int t = 0; t < L; ++t) {
    dec_pre_kernel<<<(B * H) / 256, 256, 0, stream>>>(ybuf, dec_emb_W, dec_emb_b, h_bf,
                                                      Acat1, Acat2, t);
    gemm_wmma_kernel<true, 0, EPI_F32><<<16, 256, 0, stream>>>(
        Acat1, 256, attn_W, attn_b, logits, nullptr, L, B, L, 256, 2);
    attn_apply_kernel<<<B, 256, 0, stream>>>(logits, attn_mask, enc_bf, Acat2, t);
    gemm_wmma_kernel<true, 0, EPI_TANH_BF16><<<8, 256, 0, stream>>>(
        Acat2, 256, comb_W, comb_b, nullptr, g_bf, H, B, H, 256, 1);
    gru_dual_gemm_kernel<<<dim3(24, 2), 256, 0, stream>>>(
        g_bf, (long)H, dec_Wih, dec_bih,
        h_bf, (long)H, dec_Whh, dec_bhh, gi, gh);
    gru_gates_kernel<<<(B * H) / 256, 256, 0, stream>>>(gi, gh, h_f32, h_bf, nullptr, t);
    dec_out_kernel<<<B / 8, 256, 0, stream>>>(h_f32, out_W, out_b, ybuf, outp, t);
  }
}
